// LSTMModel_81088982548520
// MI455X (gfx1250) — matrix operand order repaired and checked by reading, not compiled, not run on hardware
//
#include <hip/hip_runtime.h>

// ---- problem dims ----
#define BB      32768
#define TT      64
#define FF      8
#define HH      16
#define FC1_N   512
#define FC2_N   32
#define TILE    16          // batch rows per wave
#define NW      8           // waves per block
#define NTHREADS (NW * 32)

typedef _Float16 v16h  __attribute__((ext_vector_type(16)));
typedef _Float16 f16x8 __attribute__((ext_vector_type(8)));
typedef float    v8f   __attribute__((ext_vector_type(8)));
typedef float    f32x4 __attribute__((ext_vector_type(4)));

__device__ __forceinline__ v8f wmma16x16x32(v16h a, v16h b, v8f c) {
  // D = A(16x32 f16) x B(32x16 f16) + C(16x16 f32)
  return __builtin_amdgcn_wmma_f32_16x16x32_f16(false, a, false, b,
                                                (short)0, c, false, false);
}

__device__ __forceinline__ v8f splat8(float v) {
  v8f c;
#pragma unroll
  for (int i = 0; i < 8; ++i) c[i] = v;
  return c;
}

// sigmoid via raw HW transcendentals: v_exp_f32 + v_rcp_f32
// (avoids the precise-division v_div_scale/v_div_fmas expansion)
__device__ __forceinline__ float sigmoidf_fast(float x) {
  float e = __builtin_amdgcn_exp2f(-1.442695040888963f * x);
  return __builtin_amdgcn_rcpf(1.0f + e);
}

// Store a C/D-layout tile (lane = col n, vgpr r = row r+8*half) into a
// row-major 16x16 f16 LDS tile.
__device__ __forceinline__ void store_tile(_Float16* st, v8f h, int n, int half) {
#pragma unroll
  for (int r = 0; r < 8; ++r)
    st[(r + 8 * half) * 16 + n] = (_Float16)h[r];
}

// Load 8 consecutive f16 from row r of a 16-wide LDS tile (16B aligned).
__device__ __forceinline__ f16x8 ld_row(const _Float16* st, int r, int off) {
  return *(const f16x8*)(st + r * 16 + off);
}

// A-fragment: element i<8 -> K = 8*half + i ; i>=8 -> K = 16 + 8*half + (i-8)
__device__ __forceinline__ v16h pack_a(f16x8 lo, f16x8 hi) {
  v16h a;
#pragma unroll
  for (int i = 0; i < 8; ++i) { a[i] = lo[i]; a[i + 8] = hi[i]; }
  return a;
}

// B-fragment (32x16): element i<8 <-> K = 8*half + i ; i>=8 <-> K = 16 + 8*half + (i-8),
// column = n.
__device__ __forceinline__ v16h load_b_full(const _Float16* w, int ldn, int col, int half) {
  v16h b;
  const int kb = 16 * half;
#pragma unroll
  for (int i = 0; i < 16; ++i) b[i] = w[((i < 8 ? 8 * half : 8 + 8 * half) + i) * ldn + col];
  return b;
}

// B-fragment whose rows K=16..31 are zero (K-dim only 16 valid rows).
__device__ __forceinline__ v16h load_b_k16(const _Float16* w, int ldn, int krow0,
                                           int col, int half) {
  v16h b;
#pragma unroll
  for (int i = 0; i < 16; ++i)
    b[i] = (i >= 8) ? (_Float16)0.0f : w[(krow0 + 8 * half + i) * ldn + col];
  return b;
}

__device__ __forceinline__ void lstm_update(v8f zi, v8f zf, v8f zg, v8f zo,
                                            v8f& c, v8f& h) {
#pragma unroll
  for (int r = 0; r < 8; ++r) {
    float is = sigmoidf_fast(zi[r]);
    float fs = sigmoidf_fast(zf[r]);
    float os = sigmoidf_fast(zo[r]);
    float g  = fmaxf(zg[r], 0.0f);          // activation='relu'
    float cn = fs * c[r] + is * g;
    c[r] = cn;
    h[r] = os * fmaxf(cn, 0.0f);            // h = o * relu(c)
  }
}

__global__ __launch_bounds__(NTHREADS)
void lstm_fused_wmma(const float* __restrict__ x,
                     const float* __restrict__ W1, const float* __restrict__ U1,
                     const float* __restrict__ b1,
                     const float* __restrict__ W2, const float* __restrict__ U2,
                     const float* __restrict__ b2,
                     const float* __restrict__ Wfc1, const float* __restrict__ bfc1,
                     const float* __restrict__ Wfc2, const float* __restrict__ bfc2,
                     float* __restrict__ out) {
  // f16 weight staging (one pass per block, L2-resident source)
  __shared__ _Float16 sWU1[32 * 64];        // rows 0-7: W1, 8-23: U1, 24-31: 0
  __shared__ _Float16 sWU2[32 * 64];        // rows 0-15: W2, 16-31: U2
  __shared__ _Float16 sF1[16 * FC1_N];      // Wfc1 row-major [16][512]
  __shared__ _Float16 sF2[FC1_N * FC2_N];   // Wfc2 row-major [512][32]
  __shared__ _Float16 sStage[NW * 512];     // per-wave 2x(16x16) f16 transpose tiles

  const int tid = threadIdx.x;
  for (int i = tid; i < 32 * 64; i += NTHREADS) {
    const int k = i >> 6, col = i & 63;
    sWU1[i] = (_Float16)((k < 8) ? W1[k * 64 + col]
                                 : (k < 24 ? U1[(k - 8) * 64 + col] : 0.0f));
    sWU2[i] = (_Float16)((k < 16) ? W2[k * 64 + col] : U2[(k - 16) * 64 + col]);
  }
  for (int i = tid; i < 16 * FC1_N; i += NTHREADS) sF1[i] = (_Float16)Wfc1[i];
  for (int i = tid; i < FC1_N * FC2_N; i += NTHREADS) sF2[i] = (_Float16)Wfc2[i];
  __syncthreads();

  const int wave = tid >> 5;
  const int lane = tid & 31;
  const int half = lane >> 4;       // 0: K 0-7 / 16-23 ; 1: K 8-15 / 24-31
  const int n    = lane & 15;       // A row index / C,D column index
  _Float16* st1 = sStage + wave * 512;   // h1 transpose tile
  _Float16* st2 = st1 + 256;             // h2 transpose tile
  const int tileBase = (blockIdx.x * NW + wave) * TILE;

  // Preload LSTM B-fragments (8 x 8 VGPRs) + bias C-operands (pre-splatted).
  v16h BU1[4], BU2[4];
  v8f cb1[4], cb2[4];
#pragma unroll
  for (int g = 0; g < 4; ++g) {
    BU1[g] = load_b_full(sWU1, 64, g * 16 + n, half);
    BU2[g] = load_b_full(sWU2, 64, g * 16 + n, half);
    cb1[g] = splat8(b1[g * 16 + n]);
    cb2[g] = splat8(b2[g * 16 + n]);
  }

  v8f h1 = {}, c1 = {}, h2 = {}, c2 = {};
  const float* xrow = x + (size_t)(tileBase + n) * TT * FF;  // used by half==0 lanes
  const f16x8 zero8 = {};

  // st1 must hold h1 (initially zero) before the first layer-1 read; inside
  // the loop, layer 2 keeps st1 up to date, so layer 1 never re-stores it.
  store_tile(st1, h1, n, half);
  asm volatile("s_wait_dscnt 0" ::: "memory");

#pragma unroll 1
  for (int t = 0; t < TT; ++t) {
    // ---- layer 1: z = [x_t, h1, 0] @ [W1;U1;0] + b1 ----
    // (st1 already holds h1 from previous iteration's layer-2 store)
    f16x8 lo, hi;
    if (half == 0) {
      const float* xp = xrow + t * FF;       // 32B, 32B-aligned
      f32x4 xa = *(const f32x4*)xp;
      f32x4 xb = *(const f32x4*)(xp + 4);
#pragma unroll
      for (int i = 0; i < 4; ++i) { lo[i] = (_Float16)xa[i]; lo[i + 4] = (_Float16)xb[i]; }
      __builtin_prefetch(xp + 8 * FF, 0, 0); // 8 timesteps ahead
      hi = ld_row(st1, n, 8);                // K16-23 = h1(row,8..15)
    } else {
      lo = ld_row(st1, n, 0);                // K8-15  = h1(row,0..7)
      hi = zero8;                            // K24-31 = pad
    }
    v16h a1 = pack_a(lo, hi);
    v8f zi = wmma16x16x32(a1, BU1[0], cb1[0]);
    v8f zf = wmma16x16x32(a1, BU1[1], cb1[1]);
    v8f zg = wmma16x16x32(a1, BU1[2], cb1[2]);
    v8f zo = wmma16x16x32(a1, BU1[3], cb1[3]);
    lstm_update(zi, zf, zg, zo, c1, h1);

    // ---- layer 2: z = [h1_t, h2] @ [W2;U2] + b2 ----
    store_tile(st1, h1, n, half);            // h1(t) (also feeds next layer-1)
    store_tile(st2, h2, n, half);            // h2(t-1)
    asm volatile("s_wait_dscnt 0" ::: "memory");
    lo = ld_row(st1, n, 8 * half);           // K(0..15)  = h1
    hi = ld_row(st2, n, 8 * half);           // K(16..31) = h2
    v16h a2 = pack_a(lo, hi);
    zi = wmma16x16x32(a2, BU2[0], cb2[0]);
    zf = wmma16x16x32(a2, BU2[1], cb2[1]);
    zg = wmma16x16x32(a2, BU2[2], cb2[2]);
    zo = wmma16x16x32(a2, BU2[3], cb2[3]);
    lstm_update(zi, zf, zg, zo, c2, h2);
  }

  // ---- FC head: out = relu(h2 @ Wfc1 + bfc1) @ Wfc2 + bfc2 ----
  store_tile(st2, h2, n, half);
  asm volatile("s_wait_dscnt 0" ::: "memory");
  v16h a_h2 = pack_a(ld_row(st2, n, 8 * half), zero8);  // K=0..15 valid

  v8f out0 = splat8(bfc2[n]);
  v8f out1 = splat8(bfc2[16 + n]);

#pragma unroll 1
  for (int j = 0; j < FC1_N / 16; ++j) {
    v16h bf1 = load_b_k16(sF1, FC1_N, 0, j * 16 + n, half);
    v8f tj = wmma16x16x32(a_h2, bf1, splat8(bfc1[j * 16 + n]));
#pragma unroll
    for (int r = 0; r < 8; ++r) tj[r] = fmaxf(tj[r], 0.0f);
    store_tile(st1, tj, n, half);                       // D-layout -> A-layout
    asm volatile("s_wait_dscnt 0" ::: "memory");
    v16h at  = pack_a(ld_row(st1, n, 8 * half), zero8);
    v16h b20 = load_b_k16(sF2, FC2_N, j * 16, n, half);
    v16h b21 = load_b_k16(sF2, FC2_N, j * 16, 16 + n, half);
    out0 = wmma16x16x32(at, b20, out0);
    out1 = wmma16x16x32(at, b21, out1);
  }

#pragma unroll
  for (int r = 0; r < 8; ++r) {
    const int m = r + 8 * half;
    out[(size_t)(tileBase + m) * FC2_N + n]      = out0[r];
    out[(size_t)(tileBase + m) * FC2_N + 16 + n] = out1[r];
  }
}

extern "C" void kernel_launch(void* const* d_in, const int* in_sizes, int n_in,
                              void* d_out, int out_size, void* d_ws, size_t ws_size,
                              hipStream_t stream) {
  const float* x    = (const float*)d_in[0];
  const float* W1   = (const float*)d_in[1];
  const float* U1   = (const float*)d_in[2];
  const float* b1   = (const float*)d_in[3];
  const float* W2   = (const float*)d_in[4];
  const float* U2   = (const float*)d_in[5];
  const float* b2   = (const float*)d_in[6];
  const float* Wfc1 = (const float*)d_in[7];
  const float* bfc1 = (const float*)d_in[8];
  const float* Wfc2 = (const float*)d_in[9];
  const float* bfc2 = (const float*)d_in[10];
  float* out = (float*)d_out;

  dim3 grid(BB / (TILE * NW));   // 2048 batch tiles, 8 waves/block -> 256 blocks
  dim3 block(NTHREADS);          // 8 wave32
  lstm_fused_wmma<<<grid, block, 0, stream>>>(x, W1, U1, b1, W2, U2, b2,
                                              Wfc1, bfc1, Wfc2, bfc2, out);
}
